// MultiHeadAttention_34892314313033
// MI455X (gfx1250) — matrix operand order repaired and checked by reading, not compiled, not run on hardware
//
#include <hip/hip_runtime.h>
#include <hip/hip_bf16.h>

typedef __attribute__((ext_vector_type(16))) _Float16 v16h;
typedef __attribute__((ext_vector_type(8)))  _Float16 v8h;
typedef __attribute__((ext_vector_type(8)))  float    v8f;
typedef __attribute__((ext_vector_type(4)))  unsigned int v4u;
typedef __attribute__((ext_vector_type(8)))  int      v8i;
typedef __attribute__((ext_vector_type(4)))  int      v4i;

// ---------------------------------------------------------------- constants
#define NTOK   16384          // B*L*H*W = 2*32*16*16
#define DMODEL 512
#define D3     1536
#define NHEADS 8
#define HDIM   64
#define NGROUP 4096           // B*H*W*NHEADS
#define LDQ    72             // padded LDS stride for 64-wide q/k/v tiles (TDM pad)
#define LDP    40             // padded LDS stride for 32-wide prob tile
#define PI_F   3.14159265358979323846f

// ---------------------------------------------------------------- WMMA core
__device__ __forceinline__ v8f wmma16(v16h a, v16h b, v8f c) {
  return __builtin_amdgcn_wmma_f32_16x16x32_f16(
      /*neg_a=*/false, a, /*neg_b=*/false, b,
      /*c_mod=*/(short)0, c, /*reuse_a=*/false, /*reuse_b=*/false);
}

__device__ __forceinline__ v16h cat8(v8h a, v8h b) {
  return __builtin_shufflevector(a, b, 0,1,2,3,4,5,6,7,8,9,10,11,12,13,14,15);
}

// A fragment (16x32 f16): lane<16 -> M=lane, K {k0..k0+15} across v0-7; lane>=16 -> K base +16.
__device__ __forceinline__ v16h load_fragA(const _Float16* p, int r0, int k0, int ld) {
  const int lane = threadIdx.x & 31;
  const int r  = r0 + (lane & 15);
  const int kb = k0 + ((lane >> 4) << 4);
  const _Float16* q = p + r * ld + kb;
  v8h lo = *reinterpret_cast<const v8h*>(q);
  v8h hi = *reinterpret_cast<const v8h*>(q + 8);
  return cat8(lo, hi);
}

// B fragment (32x16 f16) from source where row n holds column n of B contiguous in K:
// lane<16 -> N=lane, K {k0..k0+15} across v0-7; lane>=16 -> K base +16.
__device__ __forceinline__ v16h load_fragB(const _Float16* p, int n0, int k0, int ld) {
  const int lane = threadIdx.x & 31;
  const int n  = n0 + (lane & 15);
  const int kb = k0 + ((lane >> 4) << 4);
  const _Float16* q = p + n * ld + kb;
  v8h lo = *reinterpret_cast<const v8h*>(q);
  v8h hi = *reinterpret_cast<const v8h*>(q + 8);
  return cat8(lo, hi);
}

// B fragment from row-major K x N source (column gather), for P@V.
__device__ __forceinline__ v16h load_fragB_col(const _Float16* p, int k0, int c0, int ld) {
  const int lane = threadIdx.x & 31;
  const int c  = c0 + (lane & 15);
  const int kb = k0 + ((lane >> 4) << 4);
  union { _Float16 h[16]; v16h v; } t;
#pragma unroll
  for (int j = 0; j < 16; ++j) t.h[j] = p[(kb + j) * ld + c];
  return t.v;
}

// ---------------------------------------------------------------- TDM staging
// 2D f16 tile load via Tensor Data Mover: 32 rows x 64 cols, row stride
// `row_stride_elems` in global, written to LDS with +16B pad per 128B row
// (pad_interval=32 DWORDs, pad_amount=4 DWORDs) -> LDS stride 72 halves.
__device__ __forceinline__ void tdm_load_tile_32x64(const _Float16* gsrc,
                                                    unsigned lds_byte,
                                                    unsigned long long row_stride_elems) {
  unsigned long long ga = (unsigned long long)(uintptr_t)gsrc;
  v4u g0 = { 1u,                                            // count=1, user desc
             lds_byte,                                      // lds_addr
             (unsigned)(ga & 0xffffffffu),                  // global_addr[31:0]
             (unsigned)((ga >> 32) & 0x01ffffffu) | (2u << 30) }; // addr[56:32], type=2
  const unsigned td0 = 64u, td1 = 32u;                      // tensor dims (no OOB)
  v8i g1;
  g1[0] = (int)((1u << 16) | (1u << 20) | (4u << 22) | (3u << 25));
          // data_size=2B | pad_enable | pad_interval: 32 DW | pad_amount: 4 DW
  g1[1] = (int)(td0 << 16);                                 // tensor_dim0[15:0]
  g1[2] = (int)((td0 >> 16) | (td1 << 16));                 // dim0 hi | tensor_dim1 lo
  g1[3] = (int)((td1 >> 16) | (64u << 16));                 // dim1 hi | tile_dim0=64
  g1[4] = (int)(32u);                                       // tile_dim1=32, tile_dim2=0
  g1[5] = (int)(row_stride_elems & 0xffffffffu);            // tensor_dim0_stride lo
  g1[6] = (int)((row_stride_elems >> 32) & 0xffffu);        // stride0 hi | stride1 lo=0
  g1[7] = 0;
  v4i z4 = {0, 0, 0, 0};
  v8i z8 = {0, 0, 0, 0, 0, 0, 0, 0};
  // 6-arg variant (clang-23 / therock-10.0): groups 2/3 disabled, extra group, cpol=0
  __builtin_amdgcn_tensor_load_to_lds(g0, g1, z4, z4, z8, 0);
}

// ---------------------------------------------------------------- kernels
__global__ void rope_cvt_f32_to_f16(const float* __restrict__ in,
                                    _Float16* __restrict__ out, int n) {
  int i = blockIdx.x * blockDim.x + threadIdx.x;
  if (i < n) out[i] = (_Float16)in[i];
}

// C(f16) = A(MxK) * B^T; one wave -> 32x64 C tile, double-buffered fragments.
__global__ __launch_bounds__(128) void rope_gemm_f16(
    const _Float16* __restrict__ A, const _Float16* __restrict__ B,
    _Float16* __restrict__ C, int K, int lda, int ldb, int ldc) {
  const int m0 = blockIdx.x * 32;
  const int n0 = (blockIdx.y * blockDim.y + threadIdx.y) * 64;
  v8f acc[2][4] = {};
  v16h a0 = load_fragA(A, m0,      0, lda);
  v16h a1 = load_fragA(A, m0 + 16, 0, lda);
  v16h b0 = load_fragB(B, n0,      0, ldb);
  v16h b1 = load_fragB(B, n0 + 16, 0, ldb);
  v16h b2 = load_fragB(B, n0 + 32, 0, ldb);
  v16h b3 = load_fragB(B, n0 + 48, 0, ldb);
  for (int k0 = 32; k0 < K; k0 += 32) {
    // issue next-k loads before consuming current fragments (double buffer)
    v16h na0 = load_fragA(A, m0,      k0, lda);
    v16h na1 = load_fragA(A, m0 + 16, k0, lda);
    v16h nb0 = load_fragB(B, n0,      k0, ldb);
    v16h nb1 = load_fragB(B, n0 + 16, k0, ldb);
    v16h nb2 = load_fragB(B, n0 + 32, k0, ldb);
    v16h nb3 = load_fragB(B, n0 + 48, k0, ldb);
    acc[0][0] = wmma16(a0, b0, acc[0][0]);
    acc[0][1] = wmma16(a0, b1, acc[0][1]);
    acc[0][2] = wmma16(a0, b2, acc[0][2]);
    acc[0][3] = wmma16(a0, b3, acc[0][3]);
    acc[1][0] = wmma16(a1, b0, acc[1][0]);
    acc[1][1] = wmma16(a1, b1, acc[1][1]);
    acc[1][2] = wmma16(a1, b2, acc[1][2]);
    acc[1][3] = wmma16(a1, b3, acc[1][3]);
    a0 = na0; a1 = na1; b0 = nb0; b1 = nb1; b2 = nb2; b3 = nb3;
  }
  acc[0][0] = wmma16(a0, b0, acc[0][0]);
  acc[0][1] = wmma16(a0, b1, acc[0][1]);
  acc[0][2] = wmma16(a0, b2, acc[0][2]);
  acc[0][3] = wmma16(a0, b3, acc[0][3]);
  acc[1][0] = wmma16(a1, b0, acc[1][0]);
  acc[1][1] = wmma16(a1, b1, acc[1][1]);
  acc[1][2] = wmma16(a1, b2, acc[1][2]);
  acc[1][3] = wmma16(a1, b3, acc[1][3]);

  const int lane = threadIdx.x & 31;
  const int col  = n0 + (lane & 15);
#pragma unroll
  for (int mt = 0; mt < 2; ++mt) {
    const int row = m0 + mt * 16 + ((lane >> 4) << 3);
#pragma unroll
    for (int t = 0; t < 4; ++t)
#pragma unroll
      for (int i = 0; i < 8; ++i)
        C[(size_t)(row + i) * ldc + col + 16 * t] = (_Float16)acc[mt][t][i];
  }
}

// Same, f32 output with bias (final projection).
__global__ __launch_bounds__(128) void rope_gemm_f32_bias(
    const _Float16* __restrict__ A, const _Float16* __restrict__ B,
    const float* __restrict__ bias, float* __restrict__ C,
    int K, int lda, int ldb, int ldc) {
  const int m0 = blockIdx.x * 32;
  const int n0 = (blockIdx.y * blockDim.y + threadIdx.y) * 64;
  v8f acc[2][4] = {};
  v16h a0 = load_fragA(A, m0,      0, lda);
  v16h a1 = load_fragA(A, m0 + 16, 0, lda);
  v16h b0 = load_fragB(B, n0,      0, ldb);
  v16h b1 = load_fragB(B, n0 + 16, 0, ldb);
  v16h b2 = load_fragB(B, n0 + 32, 0, ldb);
  v16h b3 = load_fragB(B, n0 + 48, 0, ldb);
  for (int k0 = 32; k0 < K; k0 += 32) {
    v16h na0 = load_fragA(A, m0,      k0, lda);
    v16h na1 = load_fragA(A, m0 + 16, k0, lda);
    v16h nb0 = load_fragB(B, n0,      k0, ldb);
    v16h nb1 = load_fragB(B, n0 + 16, k0, ldb);
    v16h nb2 = load_fragB(B, n0 + 32, k0, ldb);
    v16h nb3 = load_fragB(B, n0 + 48, k0, ldb);
    acc[0][0] = wmma16(a0, b0, acc[0][0]);
    acc[0][1] = wmma16(a0, b1, acc[0][1]);
    acc[0][2] = wmma16(a0, b2, acc[0][2]);
    acc[0][3] = wmma16(a0, b3, acc[0][3]);
    acc[1][0] = wmma16(a1, b0, acc[1][0]);
    acc[1][1] = wmma16(a1, b1, acc[1][1]);
    acc[1][2] = wmma16(a1, b2, acc[1][2]);
    acc[1][3] = wmma16(a1, b3, acc[1][3]);
    a0 = na0; a1 = na1; b0 = nb0; b1 = nb1; b2 = nb2; b3 = nb3;
  }
  acc[0][0] = wmma16(a0, b0, acc[0][0]);
  acc[0][1] = wmma16(a0, b1, acc[0][1]);
  acc[0][2] = wmma16(a0, b2, acc[0][2]);
  acc[0][3] = wmma16(a0, b3, acc[0][3]);
  acc[1][0] = wmma16(a1, b0, acc[1][0]);
  acc[1][1] = wmma16(a1, b1, acc[1][1]);
  acc[1][2] = wmma16(a1, b2, acc[1][2]);
  acc[1][3] = wmma16(a1, b3, acc[1][3]);

  const int lane = threadIdx.x & 31;
  const int col  = n0 + (lane & 15);
#pragma unroll
  for (int mt = 0; mt < 2; ++mt) {
    const int row = m0 + mt * 16 + ((lane >> 4) << 3);
#pragma unroll
    for (int t = 0; t < 4; ++t)
#pragma unroll
      for (int i = 0; i < 8; ++i)
        C[(size_t)(row + i) * ldc + col + 16 * t] = acc[mt][t][i] + bias[col + 16 * t];
  }
}

// Axial rotary on q/k channels [0,96).
__global__ void rope_rotary(_Float16* __restrict__ qkv) {
  int idx = blockIdx.x * blockDim.x + threadIdx.x;
  if (idx >= NTOK * 48) return;
  int t = idx / 48, p = idx % 48;
  int w = t & 15, h = (t >> 4) & 15, l = (t >> 8) & 31;
  int jj, pos, denom;
  if (p < 16)      { jj = p;      pos = l; denom = 31; }
  else if (p < 32) { jj = p - 16; pos = h; denom = 15; }
  else             { jj = p - 32; pos = w; denom = 15; }
  float tn   = -1.0f + 2.0f * (float)pos / (float)denom;
  float base = PI_F * (1.0f + (127.0f / 15.0f) * (float)jj);
  float f  = tn * base;
  float cs = __cosf(f), sn = __sinf(f);
  _Float16* qp = qkv + (size_t)t * D3 + 2 * p;
  float q0 = (float)qp[0], q1 = (float)qp[1];
  qp[0] = (_Float16)(q0 * cs - q1 * sn);
  qp[1] = (_Float16)(q1 * cs + q0 * sn);
  _Float16* kp = qp + DMODEL;
  float k0 = (float)kp[0], k1 = (float)kp[1];
  kp[0] = (_Float16)(k0 * cs - k1 * sn);
  kp[1] = (_Float16)(k1 * cs + k0 * sn);
}

// One wave per (b,h,w,head) group: 32x32x64 causal attention.
// q/k/v staged to LDS by the Tensor Data Mover (3 descriptors, 1 tensorcnt wait).
__global__ __launch_bounds__(32) void rope_attention(
    const _Float16* __restrict__ qkv, _Float16* __restrict__ out) {
  __shared__ _Float16 sq[32 * LDQ];
  __shared__ _Float16 sk[32 * LDQ];
  __shared__ _Float16 sv[32 * LDQ];
  __shared__ _Float16 sp[32 * LDP];

  const int g     = blockIdx.x;
  const int head  = g & (NHEADS - 1);
  const int spat  = g >> 3;
  const int b     = spat >> 8;
  const int hw    = spat & 255;
  const int tbase = b * 8192 + hw;        // token(l) = tbase + l*256
  const int lane  = threadIdx.x;
  const _Float16* src = qkv + (size_t)tbase * D3 + head * HDIM;

  // TDM: three 32x64 tiles (row stride 256 tokens * 1536 elems), padded into LDS.
  tdm_load_tile_32x64(src,              (unsigned)(uintptr_t)sq, 256ull * D3);
  tdm_load_tile_32x64(src + DMODEL,     (unsigned)(uintptr_t)sk, 256ull * D3);
  tdm_load_tile_32x64(src + 2 * DMODEL, (unsigned)(uintptr_t)sv, 256ull * D3);
  __builtin_amdgcn_s_wait_tensorcnt(0);
  asm volatile("" ::: "memory");
  __syncthreads();

  // S = Q * K^T : 2x2 tiles, two K chunks.
  v8f s[2][2] = {};
#pragma unroll
  for (int kc = 0; kc < 64; kc += 32) {
    v16h a0 = load_fragA(sq, 0,  kc, LDQ);
    v16h a1 = load_fragA(sq, 16, kc, LDQ);
    v16h b0 = load_fragB(sk, 0,  kc, LDQ);
    v16h b1 = load_fragB(sk, 16, kc, LDQ);
    s[0][0] = wmma16(a0, b0, s[0][0]);
    s[0][1] = wmma16(a0, b1, s[0][1]);
    s[1][0] = wmma16(a1, b0, s[1][0]);
    s[1][1] = wmma16(a1, b1, s[1][1]);
  }

  const float scale = 0.125f;             // 1/sqrt(64)
  const int colb = lane & 15;
  const int rowb = (lane >> 4) << 3;

#pragma unroll
  for (int mt = 0; mt < 2; ++mt)
#pragma unroll
    for (int nt = 0; nt < 2; ++nt)
#pragma unroll
      for (int i = 0; i < 8; ++i) {
        int r = mt * 16 + rowb + i, c = nt * 16 + colb;
        float v = s[mt][nt][i] * scale;
        s[mt][nt][i] = (c <= r) ? v : -1.0e30f;
      }

  // row softmax: reductions stay within 16-lane halves (rows preserved).
#pragma unroll
  for (int mt = 0; mt < 2; ++mt) {
    float mx[8], sm[8];
#pragma unroll
    for (int i = 0; i < 8; ++i) mx[i] = fmaxf(s[mt][0][i], s[mt][1][i]);
#pragma unroll
    for (int off = 1; off < 16; off <<= 1)
#pragma unroll
      for (int i = 0; i < 8; ++i) mx[i] = fmaxf(mx[i], __shfl_xor(mx[i], off, 32));
#pragma unroll
    for (int i = 0; i < 8; ++i) {
      s[mt][0][i] = __expf(s[mt][0][i] - mx[i]);
      s[mt][1][i] = __expf(s[mt][1][i] - mx[i]);
      sm[i] = s[mt][0][i] + s[mt][1][i];
    }
#pragma unroll
    for (int off = 1; off < 16; off <<= 1)
#pragma unroll
      for (int i = 0; i < 8; ++i) sm[i] += __shfl_xor(sm[i], off, 32);
#pragma unroll
    for (int i = 0; i < 8; ++i) {
      float rinv = 1.0f / sm[i];
      s[mt][0][i] *= rinv;
      s[mt][1][i] *= rinv;
    }
#pragma unroll
    for (int nt = 0; nt < 2; ++nt)
#pragma unroll
      for (int i = 0; i < 8; ++i)
        sp[(mt * 16 + rowb + i) * LDP + nt * 16 + colb] = (_Float16)s[mt][nt][i];
  }
  __syncthreads();

  // O = P * V : M=32, N=64, K=32.
  v16h pa0 = load_fragA(sp, 0,  0, LDP);
  v16h pa1 = load_fragA(sp, 16, 0, LDP);
  v8f o[2][4] = {};
#pragma unroll
  for (int nt = 0; nt < 4; ++nt) {
    v16h bv = load_fragB_col(sv, 0, nt * 16, LDQ);
    o[0][nt] = wmma16(pa0, bv, o[0][nt]);
    o[1][nt] = wmma16(pa1, bv, o[1][nt]);
  }
#pragma unroll
  for (int mt = 0; mt < 2; ++mt)
#pragma unroll
    for (int nt = 0; nt < 4; ++nt)
#pragma unroll
      for (int i = 0; i < 8; ++i) {
        int l = mt * 16 + rowb + i;
        out[(size_t)(tbase + l * 256) * DMODEL + head * HDIM + nt * 16 + colb] =
            (_Float16)o[mt][nt][i];
      }
}

// ---------------------------------------------------------------- launcher
extern "C" void kernel_launch(void* const* d_in, const int* in_sizes, int n_in,
                              void* d_out, int out_size, void* d_ws, size_t ws_size,
                              hipStream_t stream) {
  const float* x    = (const float*)d_in[0];
  const float* Wqkv = (const float*)d_in[1];
  const float* Wfc  = (const float*)d_in[2];
  const float* bfc  = (const float*)d_in[3];
  float* out = (float*)d_out;

  _Float16* xh    = (_Float16*)d_ws;
  _Float16* wqkvh = xh    + (size_t)NTOK * DMODEL;
  _Float16* wfch  = wqkvh + (size_t)D3 * DMODEL;
  _Float16* qkvh  = wfch  + (size_t)DMODEL * DMODEL;
  _Float16* aoh   = qkvh  + (size_t)NTOK * D3;

  rope_cvt_f32_to_f16<<<(NTOK * DMODEL + 255) / 256, 256, 0, stream>>>(x, xh, NTOK * DMODEL);
  rope_cvt_f32_to_f16<<<(D3 * DMODEL + 255) / 256, 256, 0, stream>>>(Wqkv, wqkvh, D3 * DMODEL);
  rope_cvt_f32_to_f16<<<(DMODEL * DMODEL + 255) / 256, 256, 0, stream>>>(Wfc, wfch, DMODEL * DMODEL);

  rope_gemm_f16<<<dim3(NTOK / 32, D3 / 64 / 4), dim3(32, 4), 0, stream>>>(
      xh, wqkvh, qkvh, DMODEL, DMODEL, DMODEL, D3);

  rope_rotary<<<(NTOK * 48 + 255) / 256, 256, 0, stream>>>(qkvh);

  rope_attention<<<NGROUP, 32, 0, stream>>>(qkvh, aoh);

  rope_gemm_f32_bias<<<dim3(NTOK / 32, DMODEL / 64 / 4), dim3(32, 4), 0, stream>>>(
      aoh, wfch, bfc, out, DMODEL, DMODEL, DMODEL, DMODEL);

  (void)in_sizes; (void)n_in; (void)out_size; (void)ws_size;
}
